// GNNDenseLayer_36919538876772
// MI455X (gfx1250) — matrix operand order repaired and checked by reading, not compiled, not run on hardware
//
#include <hip/hip_runtime.h>

// CDNA5 / gfx1250 wave32 WMMA types
typedef __attribute__((ext_vector_type(16))) __bf16 v16bf;
typedef __attribute__((ext_vector_type(8)))  float  v8f;

#define BN_EPS 1e-5f

// ---------------------------------------------------------------------------
// Utility: grid-stride f32 zero (graph-capture-safe replacement for memset)
// ---------------------------------------------------------------------------
__global__ void zero_f32(float* __restrict__ p, long long n) {
  long long i = (long long)blockIdx.x * blockDim.x + threadIdx.x;
  long long stride = (long long)gridDim.x * blockDim.x;
  for (; i < n; i += stride) p[i] = 0.0f;
}

// ---------------------------------------------------------------------------
// Weight prep: f32 [K x Ncol] (row-major, as used by x @ W) -> bf16 [Ncol x K]
// Transposed layout makes the WMMA B-operand a single contiguous 32B load.
// ---------------------------------------------------------------------------
__global__ void prep_wt(const float* __restrict__ W, __bf16* __restrict__ Wt,
                        int K, int Ncol) {
  int idx = blockIdx.x * blockDim.x + threadIdx.x;
  if (idx < K * Ncol) {
    int n = idx / K;
    int k = idx - n * K;
    Wt[idx] = (__bf16)W[(long long)k * Ncol + n];
  }
}

// ---------------------------------------------------------------------------
// Edge scatter-add: agg[dst[e]][:] += x[src[e]][:], feature dim = 128.
// One wave per edge; lane l handles 4 consecutive floats (float4 gather is a
// fully coalesced 512B burst per wave), then 4 f32 global atomics.
// ---------------------------------------------------------------------------
__global__ void scatter_add128(const float* __restrict__ x,
                               const int* __restrict__ src,
                               const int* __restrict__ dst,
                               float* __restrict__ agg, int E) {
  long long gid = (long long)blockIdx.x * blockDim.x + threadIdx.x;
  int e = (int)(gid >> 5);
  if (e >= E) return;
  int c = ((int)gid & 31) * 4;
  int s = src[e];
  int d = dst[e];
  const float4 v = *(const float4*)(x + (long long)s * 128 + c);
  float* p = agg + (long long)d * 128 + c;
  atomicAdd(p + 0, v.x);
  atomicAdd(p + 1, v.y);
  atomicAdd(p + 2, v.z);
  atomicAdd(p + 3, v.w);
}

// ---------------------------------------------------------------------------
// Dual-source WMMA GEMM:  out = A1 @ W1 + A2 @ W2 + bias   (K fixed at 128)
//   A1/A2 : f32 [M x 128] row-major (converted to bf16 on the fly)
//   Wt1/2 : bf16 [DOUT x 128]  (pre-transposed weights, contiguous K per col)
//   out   : f32 [M x DOUT]
// One 16-row node tile per block; wave w owns output columns [16w, 16w+16).
// Layouts used here (A/B k order is this code's choice; C/D per CDNA5 ISA §7.12.2):
//   A 16x32 bf16: lane row = lane&15, K = kq*16 + i  (contiguous, matches B)
//   B 32x16 bf16: lane col = lane&15, K = kq*16 + i  (contiguous -> v16bf ld)
//   C/D 16x16 f32: elem r -> row r + 8*kq, col lane&15
// ---------------------------------------------------------------------------
template <int DOUT>
__global__ __launch_bounds__(32 * (DOUT / 16))
void gemm_dual_wmma(const float* __restrict__ A1, const float* __restrict__ A2,
                    const __bf16* __restrict__ Wt1, const __bf16* __restrict__ Wt2,
                    const float* __restrict__ bias, float* __restrict__ out, int M) {
  const int lane = threadIdx.x & 31;
  const int wave = threadIdx.x >> 5;   // column-tile index
  const int kq   = lane >> 4;          // which half-wave
  const int lr   = lane & 15;
  const int rowA = blockIdx.x * 16 + lr;          // A row this lane loads
  const int col  = wave * 16 + lr;                // B / C column
  const int rowC = (rowA < M) ? rowA : (M - 1);   // clamp loads (EXEC stays all-1)

  v8f c;
  const float bval = bias[col];
#pragma unroll
  for (int r = 0; r < 8; ++r) c[r] = bval;

  const float*  srcs[2] = {A1, A2};
  const __bf16* wts[2]  = {Wt1, Wt2};

#pragma unroll
  for (int m = 0; m < 2; ++m) {
    const float*  Abase = srcs[m] + (long long)rowC * 128 + kq * 16;
    const __bf16* Bbase = wts[m]  + (long long)col  * 128 + kq * 16;
#pragma unroll
    for (int k0 = 0; k0 < 128; k0 += 32) {
      const float4 f0 = *(const float4*)(Abase + k0 + 0);
      const float4 f1 = *(const float4*)(Abase + k0 + 4);
      const float4 f2 = *(const float4*)(Abase + k0 + 8);
      const float4 f3 = *(const float4*)(Abase + k0 + 12);
      v16bf a;
      a[0]  = (__bf16)f0.x; a[1]  = (__bf16)f0.y; a[2]  = (__bf16)f0.z; a[3]  = (__bf16)f0.w;
      a[4]  = (__bf16)f1.x; a[5]  = (__bf16)f1.y; a[6]  = (__bf16)f1.z; a[7]  = (__bf16)f1.w;
      a[8]  = (__bf16)f2.x; a[9]  = (__bf16)f2.y; a[10] = (__bf16)f2.z; a[11] = (__bf16)f2.w;
      a[12] = (__bf16)f3.x; a[13] = (__bf16)f3.y; a[14] = (__bf16)f3.z; a[15] = (__bf16)f3.w;
      const v16bf b = *(const v16bf*)(Bbase + k0);   // contiguous 32B, aligned
      c = __builtin_amdgcn_wmma_f32_16x16x32_bf16(
              /*neg_a=*/false, a, /*neg_b=*/false, b,
              /*c_mod=*/(short)0, c, /*reuse_a=*/false, /*reuse_b=*/false);
    }
  }

#pragma unroll
  for (int r = 0; r < 8; ++r) {
    const int mrow = blockIdx.x * 16 + kq * 8 + r;
    if (mrow < M) out[(long long)mrow * DOUT + col] = c[r];
  }
}

// ---------------------------------------------------------------------------
// BatchNorm column stats: stats[0:D] = sum, stats[D:2D] = sum of squares.
// Each thread owns one column within its row-slab; register accumulation,
// then one pair of f32 atomics per thread (gridDim*256/D atomics per column).
// ---------------------------------------------------------------------------
template <int D>
__global__ void bn_stats(const float* __restrict__ h, float* __restrict__ stats, int M) {
  const int RPB = 256 / D;
  const int col = threadIdx.x % D;
  float s = 0.0f, s2 = 0.0f;
  for (int r = blockIdx.x * RPB + threadIdx.x / D; r < M; r += gridDim.x * RPB) {
    const float v = h[(long long)r * D + col];
    s += v;
    s2 += v * v;
  }
  atomicAdd(&stats[col], s);
  atomicAdd(&stats[D + col], s2);
}

// Normalize (biased var = E[x^2] - mu^2, matching the reference) + ReLU.
template <int D>
__global__ void bn_relu(const float* __restrict__ h, const float* __restrict__ stats,
                        const float* __restrict__ gamma, const float* __restrict__ beta,
                        float* __restrict__ out, int M) {
  long long idx = (long long)blockIdx.x * blockDim.x + threadIdx.x;
  const long long total  = (long long)M * D;
  const long long stride = (long long)gridDim.x * blockDim.x;
  const float invM = 1.0f / (float)M;
  for (; idx < total; idx += stride) {
    const int col  = (int)(idx % D);
    const float mu  = stats[col] * invM;
    const float var = stats[D + col] * invM - mu * mu;
    const float v = (h[idx] - mu) * rsqrtf(var + BN_EPS) * gamma[col] + beta[col];
    out[idx] = v > 0.0f ? v : 0.0f;
  }
}

// ---------------------------------------------------------------------------
// Launch: two GraphConv + BN + ReLU layers.
// ---------------------------------------------------------------------------
extern "C" void kernel_launch(void* const* d_in, const int* in_sizes, int n_in,
                              void* d_out, int out_size, void* d_ws, size_t ws_size,
                              hipStream_t stream) {
  const float* x      = (const float*)d_in[0];
  const int*   eidx   = (const int*)d_in[1];
  const float* Wrel1  = (const float*)d_in[2];
  const float* brel1  = (const float*)d_in[3];
  const float* Wroot1 = (const float*)d_in[4];
  const float* gamma1 = (const float*)d_in[5];
  const float* beta1  = (const float*)d_in[6];
  const float* Wrel2  = (const float*)d_in[7];
  const float* brel2  = (const float*)d_in[8];
  const float* Wroot2 = (const float*)d_in[9];
  const float* gamma2 = (const float*)d_in[10];
  const float* beta2  = (const float*)d_in[11];

  const int N = in_sizes[0] / 128;  // nodes
  const int E = in_sizes[1] / 2;    // edges
  const int* src = eidx;
  const int* dst = eidx + E;

  // Workspace carve-out, 256B-aligned slices.
  char* ws = (char*)d_ws;
  size_t off = 0;
  auto carve = [&](size_t bytes) -> char* {
    char* p = ws + off;
    off = (off + bytes + 255) & ~(size_t)255;
    return p;
  };
  float*  agg   = (float*)carve((size_t)N * 128 * sizeof(float));  // agg1, reused as agg2
  float*  hpre  = (float*)carve((size_t)N * 128 * sizeof(float));  // pre-BN, reused for layer 2
  float*  h     = (float*)carve((size_t)N * 128 * sizeof(float));  // post-BN layer-1 activations
  __bf16* WtR1  = (__bf16*)carve(128 * 128 * sizeof(__bf16));
  __bf16* WtO1  = (__bf16*)carve(128 * 128 * sizeof(__bf16));
  __bf16* WtR2  = (__bf16*)carve(128 * 32 * sizeof(__bf16));
  __bf16* WtO2  = (__bf16*)carve(128 * 32 * sizeof(__bf16));
  float*  stats = (float*)carve(256 * sizeof(float));
  (void)ws_size; (void)n_in; (void)out_size;

  // Weight conversion/transposition (tiny).
  prep_wt<<<(128 * 128 + 255) / 256, 256, 0, stream>>>(Wrel1,  WtR1, 128, 128);
  prep_wt<<<(128 * 128 + 255) / 256, 256, 0, stream>>>(Wroot1, WtO1, 128, 128);
  prep_wt<<<(128 * 32  + 255) / 256, 256, 0, stream>>>(Wrel2,  WtR2, 128, 32);
  prep_wt<<<(128 * 32  + 255) / 256, 256, 0, stream>>>(Wroot2, WtO2, 128, 32);

  const int tiles = (N + 15) / 16;
  const long long scatterThreads = (long long)E * 32;
  const int scatterBlocks = (int)((scatterThreads + 255) / 256);

  // ---------------- Layer 1 (128 -> 128) ----------------
  zero_f32<<<2048, 256, 0, stream>>>(agg, (long long)N * 128);
  scatter_add128<<<scatterBlocks, 256, 0, stream>>>(x, src, dst, agg, E);
  gemm_dual_wmma<128><<<tiles, 256, 0, stream>>>(agg, x, WtR1, WtO1, brel1, hpre, N);
  zero_f32<<<1, 256, 0, stream>>>(stats, 256);
  bn_stats<128><<<1024, 256, 0, stream>>>(hpre, stats, N);
  bn_relu<128><<<2048, 256, 0, stream>>>(hpre, stats, gamma1, beta1, h, N);

  // ---------------- Layer 2 (128 -> 32) ----------------
  zero_f32<<<2048, 256, 0, stream>>>(agg, (long long)N * 128);
  scatter_add128<<<scatterBlocks, 256, 0, stream>>>(h, src, dst, agg, E);
  gemm_dual_wmma<32><<<tiles, 64, 0, stream>>>(agg, h, WtR2, WtO2, brel2, hpre, N);
  zero_f32<<<1, 256, 0, stream>>>(stats, 256);
  bn_stats<32><<<1024, 256, 0, stream>>>(hpre, stats, N);
  bn_relu<32><<<2048, 256, 0, stream>>>(hpre, stats, gamma2, beta2, (float*)d_out, N);
}
